// RNN_50740743635171
// MI455X (gfx1250) — matrix operand order repaired and checked by reading, not compiled, not run on hardware
//
#include <hip/hip_runtime.h>

#define H     1024
#define NIN   128
#define NOUT  128
#define BATCH 64
#define SEQ   512
#define BT    16      // batch rows per workgroup (WMMA M)
#define NTW   4       // column tiles (of 16) per wave: 64 tiles / 16 waves
#define HSTRB 1040    // bf16 LDS row stride for hidden states (32B skew/row)
#define XSTRB 144     // bf16 LDS row stride for x stage (128 + 16)

typedef __attribute__((ext_vector_type(8)))  __bf16 v8bf;
typedef __attribute__((ext_vector_type(16))) __bf16 v16bf;
typedef __attribute__((ext_vector_type(8)))  float  v8f;

__device__ __forceinline__ v8f wmma_bf16(v16bf a, v16bf b, v8f c) {
  // D(16x16 f32) = A(16x32 bf16) * B(32x16 bf16) + C
  return __builtin_amdgcn_wmma_f32_16x16x32_bf16(
      /*neg_a=*/false, a, /*neg_b=*/false, b,
      /*c_mod=*/(short)0, c, /*reuse_a=*/false, /*reuse_b=*/false);
}

// Hardware tanh (CDNA5 V_TANH_F32 transcendental) instead of the ~20-op
// OCML polynomial: the tanh/writeback phase is serial on the recurrence
// critical path (runs 512 times between barriers).
__device__ __forceinline__ float htanh(float x) {
#if __has_builtin(__builtin_amdgcn_tanhf)
  return __builtin_amdgcn_tanhf(x);
#else
  float r;
  // v_nop covers the TRANS-op result/source hazard (1 slot) since the
  // compiler cannot see through the asm.
  asm("v_tanh_f32 %0, %1\n\tv_nop" : "=v"(r) : "v"(x));
  return r;
#endif
}

__device__ __forceinline__ v8f splat8(float s) {
  v8f v;
#pragma unroll
  for (int r = 0; r < 8; ++r) v[r] = s;
  return v;
}

// A-fragment: lane (m,half) holds K = kb+16*half..+15 (contiguous, matches B).
__device__ __forceinline__ v16bf load_a(const __bf16* rowk, int half) {
  return *(const v16bf*)(rowk + 16 * half);
}

// B-fragment: lane (n,half) holds K = kb+16*half..+15, contiguous in a
// row-major W (B[k][n] = W[n][k] gives the transpose for free).
__device__ __forceinline__ v16bf load_b(const __bf16* w) {
  return *(const v16bf*)w;
}

__device__ __forceinline__ void split1(float v, __bf16* hi, __bf16* lo) {
  const __bf16 h = (__bf16)v;
  *hi = h;
  *lo = (__bf16)(v - (float)h);
}

// bf16x3 inner product block: acc += ah*bh + al*bh + ah*bl  (~f32 fidelity)
__device__ __forceinline__ v8f x3_block(v16bf ah, v16bf al,
                                        const __bf16* wh, const __bf16* wl,
                                        v8f acc) {
  const v16bf bh = load_b(wh);
  acc = wmma_bf16(ah, bh, acc);
  const v16bf bl = load_b(wl);
  acc = wmma_bf16(al, bh, acc);
  acc = wmma_bf16(ah, bl, acc);
  return acc;
}

// ---- prep: split an f32 matrix into hi/lo bf16 planes (bf16x3 scheme) ----
__global__ void split_bf16_kernel(const float* __restrict__ src,
                                  __bf16* __restrict__ hi,
                                  __bf16* __restrict__ lo, int n) {
  const int i = blockIdx.x * blockDim.x + threadIdx.x;
  if (i < n) split1(src[i], &hi[i], &lo[i]);
}

__global__ __launch_bounds__(512, 1) void rnn_fused_kernel(
    const float* __restrict__ x,
    const __bf16* __restrict__ Wih0h, const __bf16* __restrict__ Wih0l,
    const float* __restrict__ bih0,
    const __bf16* __restrict__ Whh0h, const __bf16* __restrict__ Whh0l,
    const float* __restrict__ bhh0,
    const __bf16* __restrict__ Wih1h, const __bf16* __restrict__ Wih1l,
    const float* __restrict__ bih1,
    const __bf16* __restrict__ Whh1h, const __bf16* __restrict__ Whh1l,
    const float* __restrict__ bhh1,
    const __bf16* __restrict__ Wfch,  const __bf16* __restrict__ Wfcl,
    const float* __restrict__ bfc,
    float* __restrict__ out)
{
  __shared__ __bf16 h0hi[BT * HSTRB];
  __shared__ __bf16 h0lo[BT * HSTRB];
  __shared__ __bf16 h1hi[BT * HSTRB];
  __shared__ __bf16 h1lo[BT * HSTRB];
  __shared__ __bf16 xshi[BT * XSTRB];
  __shared__ __bf16 xslo[BT * XSTRB];

  const int tid   = threadIdx.x;
  const int wave  = tid >> 5;
  const int lane  = tid & 31;
  const int half  = lane >> 4;
  const int tn    = lane & 15;      // A row (M) and B/D column (N)
  const int koffB = 16 * half;      // B-fragment K offset for this lane
  const int bbase = blockIdx.x * BT;

  int nrow[NTW];                    // output columns owned by this wave
#pragma unroll
  for (int j = 0; j < NTW; ++j) nrow[j] = (wave * NTW + j) * 16 + tn;

  for (int i = tid; i < BT * HSTRB; i += 512) {
    h0hi[i] = (__bf16)0.0f; h0lo[i] = (__bf16)0.0f;
    h1hi[i] = (__bf16)0.0f; h1lo[i] = (__bf16)0.0f;
  }
  __syncthreads();

  for (int t = 0; t < SEQ; ++t) {
    // ---- stage + split x_t (16 x 128): wave = batch row, lane = 4 floats ----
    {
      const int row = wave;
      const int col = lane * 4;
      const float4 v =
          *(const float4*)(x + ((size_t)(bbase + row) * SEQ + t) * NIN + col);
      split1(v.x, &xshi[row * XSTRB + col + 0], &xslo[row * XSTRB + col + 0]);
      split1(v.y, &xshi[row * XSTRB + col + 1], &xslo[row * XSTRB + col + 1]);
      split1(v.z, &xshi[row * XSTRB + col + 2], &xslo[row * XSTRB + col + 2]);
      split1(v.w, &xshi[row * XSTRB + col + 3], &xslo[row * XSTRB + col + 3]);
    }
    __syncthreads();

    // =================== layer 0: tanh(x_t Wih0^T + b + h0 Whh0^T) ==========
    v8f acc[NTW];
#pragma unroll
    for (int j = 0; j < NTW; ++j) acc[j] = splat8(bih0[nrow[j]] + bhh0[nrow[j]]);

#pragma unroll 1
    for (int kb = 0; kb < NIN; kb += 32) {
      const v16bf ah = load_a(&xshi[tn * XSTRB + kb], half);
      const v16bf al = load_a(&xslo[tn * XSTRB + kb], half);
#pragma unroll
      for (int j = 0; j < NTW; ++j) {
        const size_t wo = (size_t)nrow[j] * NIN + kb + koffB;
        acc[j] = x3_block(ah, al, Wih0h + wo, Wih0l + wo, acc[j]);
      }
    }
#pragma unroll 1
    for (int kb = 0; kb < H; kb += 32) {
      const v16bf ah = load_a(&h0hi[tn * HSTRB + kb], half);
      const v16bf al = load_a(&h0lo[tn * HSTRB + kb], half);
#pragma unroll
      for (int j = 0; j < NTW; ++j) {
        const size_t wo = (size_t)nrow[j] * H + kb + koffB;
        acc[j] = x3_block(ah, al, Whh0h + wo, Whh0l + wo, acc[j]);
      }
    }
    __syncthreads();  // all reads of h0(t-1) complete
#pragma unroll
    for (int j = 0; j < NTW; ++j) {
#pragma unroll
      for (int r = 0; r < 8; ++r) {
        const int mm = r + half * 8;          // D layout: M = r + 8*half
        const float hv = htanh(acc[j][r]);
        split1(hv, &h0hi[mm * HSTRB + nrow[j]], &h0lo[mm * HSTRB + nrow[j]]);
      }
    }
    __syncthreads();  // h0 now holds step t

    // =================== layer 1: tanh(h0_t Wih1^T + b + h1 Whh1^T) =========
#pragma unroll
    for (int j = 0; j < NTW; ++j) acc[j] = splat8(bih1[nrow[j]] + bhh1[nrow[j]]);

#pragma unroll 1
    for (int kb = 0; kb < H; kb += 32) {
      const v16bf ah = load_a(&h0hi[tn * HSTRB + kb], half);
      const v16bf al = load_a(&h0lo[tn * HSTRB + kb], half);
#pragma unroll
      for (int j = 0; j < NTW; ++j) {
        const size_t wo = (size_t)nrow[j] * H + kb + koffB;
        acc[j] = x3_block(ah, al, Wih1h + wo, Wih1l + wo, acc[j]);
      }
    }
#pragma unroll 1
    for (int kb = 0; kb < H; kb += 32) {
      const v16bf ah = load_a(&h1hi[tn * HSTRB + kb], half);
      const v16bf al = load_a(&h1lo[tn * HSTRB + kb], half);
#pragma unroll
      for (int j = 0; j < NTW; ++j) {
        const size_t wo = (size_t)nrow[j] * H + kb + koffB;
        acc[j] = x3_block(ah, al, Whh1h + wo, Whh1l + wo, acc[j]);
      }
    }
    __syncthreads();  // all reads of h1(t-1) complete
#pragma unroll
    for (int j = 0; j < NTW; ++j) {
#pragma unroll
      for (int r = 0; r < 8; ++r) {
        const int mm = r + half * 8;
        const float hv = htanh(acc[j][r]);
        split1(hv, &h1hi[mm * HSTRB + nrow[j]], &h1lo[mm * HSTRB + nrow[j]]);
      }
    }
    __syncthreads();  // h1 now holds step t
  }

  // =================== FC: out = h1(S-1) @ Wfc^T + bfc ======================
  if (wave < 8) {                     // 8 waves x 16 cols = 128 outputs
    const int nr = wave * 16 + tn;
    v8f acc = splat8(bfc[nr]);
#pragma unroll 1
    for (int kb = 0; kb < H; kb += 32) {
      const v16bf ah = load_a(&h1hi[tn * HSTRB + kb], half);
      const v16bf al = load_a(&h1lo[tn * HSTRB + kb], half);
      const size_t wo = (size_t)nr * H + kb + koffB;
      acc = x3_block(ah, al, Wfch + wo, Wfcl + wo, acc);
    }
#pragma unroll
    for (int r = 0; r < 8; ++r) {
      const int mm = r + half * 8;
      out[(size_t)(bbase + mm) * NOUT + nr] = acc[r];
    }
  }
}

extern "C" void kernel_launch(void* const* d_in, const int* in_sizes, int n_in,
                              void* d_out, int out_size, void* d_ws, size_t ws_size,
                              hipStream_t stream) {
  const float* x    = (const float*)d_in[0];
  const float* Wih0 = (const float*)d_in[1];
  const float* bih0 = (const float*)d_in[2];
  const float* Whh0 = (const float*)d_in[3];
  const float* bhh0 = (const float*)d_in[4];
  const float* Wih1 = (const float*)d_in[5];
  const float* bih1 = (const float*)d_in[6];
  const float* Whh1 = (const float*)d_in[7];
  const float* bhh1 = (const float*)d_in[8];
  const float* Wfc  = (const float*)d_in[9];
  const float* bfc  = (const float*)d_in[10];

  // Workspace layout: hi/lo bf16 planes of all weight matrices (~13 MB).
  __bf16* p = (__bf16*)d_ws;
  __bf16* Wih0h = p; p += (size_t)H * NIN;
  __bf16* Wih0l = p; p += (size_t)H * NIN;
  __bf16* Whh0h = p; p += (size_t)H * H;
  __bf16* Whh0l = p; p += (size_t)H * H;
  __bf16* Wih1h = p; p += (size_t)H * H;
  __bf16* Wih1l = p; p += (size_t)H * H;
  __bf16* Whh1h = p; p += (size_t)H * H;
  __bf16* Whh1l = p; p += (size_t)H * H;
  __bf16* Wfch  = p; p += (size_t)NOUT * H;
  __bf16* Wfcl  = p; p += (size_t)NOUT * H;

  const int TPB = 256;
  split_bf16_kernel<<<(H * NIN + TPB - 1) / TPB, TPB, 0, stream>>>(
      Wih0, Wih0h, Wih0l, H * NIN);
  split_bf16_kernel<<<(H * H + TPB - 1) / TPB, TPB, 0, stream>>>(
      Whh0, Whh0h, Whh0l, H * H);
  split_bf16_kernel<<<(H * H + TPB - 1) / TPB, TPB, 0, stream>>>(
      Wih1, Wih1h, Wih1l, H * H);
  split_bf16_kernel<<<(H * H + TPB - 1) / TPB, TPB, 0, stream>>>(
      Whh1, Whh1h, Whh1l, H * H);
  split_bf16_kernel<<<(NOUT * H + TPB - 1) / TPB, TPB, 0, stream>>>(
      Wfc, Wfch, Wfcl, NOUT * H);

  rnn_fused_kernel<<<BATCH / BT, 512, 0, stream>>>(
      x, Wih0h, Wih0l, bih0, Whh0h, Whh0l, bhh0,
      Wih1h, Wih1l, bih1, Whh1h, Whh1l, bhh1,
      Wfch, Wfcl, bfc, (float*)d_out);
}
